// MessagePassing_50964081935237
// MI455X (gfx1250) — compile-verified, matrix operand order checked by reading, not run on hardware
//
#include <hip/hip_runtime.h>
#include <hip/hip_bf16.h>

// ---- problem constants (b=8, m=4096 nodes, e=2048 edges, d=128) ----
#define NB 8
#define NM 4096
#define NE 2048
#define ND 128
#define LSTR 40   // LDS row stride in bf16 elements (80B, keeps b128 reads 16B-aligned)

typedef __attribute__((ext_vector_type(16))) __bf16 v16bf;
typedef __attribute__((ext_vector_type(8)))  __bf16 v8bf;
typedef __attribute__((ext_vector_type(4)))  __bf16 v4bf;
typedef __attribute__((ext_vector_type(8)))  float  v8f;
typedef __attribute__((ext_vector_type(4)))  float  v4f;

// Load one 16x32 bf16 A/B fragment from an LDS tile (row-major, stride LSTR).
// ISA 7.12.2 16-bit 16x32 layout: lanes 0-15 hold K=0..7 (v0-3) and K=16..23 (v4-7);
// lanes 16-31 hold K=8..15 and K=24..31, for row M = lane & 15.
__device__ __forceinline__ v16bf frag_ld(const __bf16* tile, int lane) {
  const __bf16* p = tile + (lane & 15) * LSTR + ((lane >> 4) << 3);
  v8bf lo = *(const v8bf*)p;
  v8bf hi = *(const v8bf*)(p + 16);
  return __builtin_shufflevector(lo, hi, 0,1,2,3,4,5,6,7,8,9,10,11,12,13,14,15);
}

__device__ __forceinline__ v8f wmma_bf(v16bf a, v16bf b, v8f c) {
  return __builtin_amdgcn_wmma_f32_16x16x32_bf16(false, a, false, b, (short)0, c, false, false);
}

__device__ __forceinline__ v4bf cvt4(v4f v) {
  v4bf o; o[0]=(__bf16)v[0]; o[1]=(__bf16)v[1]; o[2]=(__bf16)v[2]; o[3]=(__bf16)v[3]; return o;
}

// ---------------------------------------------------------------------------
// Kernel 1: features [b][m][d] f32 -> featT [b][d][m] bf16 (tile transpose via LDS)
// ---------------------------------------------------------------------------
__global__ void __launch_bounds__(256) k_transpose_feat(const float* __restrict__ feat,
                                                        __bf16* __restrict__ featT) {
  __shared__ __align__(16) __bf16 sT[ND * 72];   // [d][m-tile], stride 72 elems
  const int tid = threadIdx.x;
  const int b = blockIdx.y;
  const int m0 = blockIdx.x * 64;
  for (int i = 0; i < 8; ++i) {                  // 64x128 floats = 2048 float4 chunks
    int idx = tid + i * 256;
    int r = idx >> 5, c4 = (idx & 31) * 4;
    v4f v = *(const v4f*)(feat + ((size_t)b * NM + m0 + r) * ND + c4);
    sT[(c4 + 0) * 72 + r] = (__bf16)v[0];
    sT[(c4 + 1) * 72 + r] = (__bf16)v[1];
    sT[(c4 + 2) * 72 + r] = (__bf16)v[2];
    sT[(c4 + 3) * 72 + r] = (__bf16)v[3];
  }
  __syncthreads();
  for (int i = 0; i < 16; ++i) {                 // 128d x 64m bf16 = 4096 dword chunks
    int idx = tid + i * 256;
    int d = idx >> 5, m2 = (idx & 31) * 2;
    *(unsigned int*)(featT + ((size_t)b * ND + d) * NM + m0 + m2) =
        *(const unsigned int*)(sT + d * 72 + m2);
  }
}

// ---------------------------------------------------------------------------
// Kernel 2 (reused 3x): C[M,128] = A_f32[M,128] @ W_f32[128,128]^T  (bf16 WMMA)
//   outF  (optional): f32 row-major [M][128]
//   outT  (optional): bf16 transposed per batch: [b][128][rowsPerBatch]
// ---------------------------------------------------------------------------
__global__ void __launch_bounds__(256) k_gemm_nt_128(const float* __restrict__ A,
                                                     const float* __restrict__ W,
                                                     float* __restrict__ outF,
                                                     __bf16* __restrict__ outT,
                                                     int rowsPerBatch) {
  __shared__ __align__(16) __bf16 sA[64 * LSTR];
  __shared__ __align__(16) __bf16 sB[ND * LSTR];
  const int tid = threadIdx.x, lane = tid & 31, wave = tid >> 5;
  const int wm = wave >> 2, wn = wave & 3;
  const int m0 = blockIdx.x * 64;

  v8f acc[2][2] = {};
  for (int k0 = 0; k0 < 128; k0 += 32) {
    for (int i = 0; i < 2; ++i) {                // A tile 64x32 f32 -> bf16
      int idx = tid + i * 256;
      int r = idx >> 3, c4 = (idx & 7) * 4;
      v4f v = *(const v4f*)(A + (size_t)(m0 + r) * 128 + k0 + c4);
      *(v4bf*)(sA + r * LSTR + c4) = cvt4(v);
    }
    for (int i = 0; i < 4; ++i) {                // W tile 128x32
      int idx = tid + i * 256;
      int r = idx >> 3, c4 = (idx & 7) * 4;
      v4f v = *(const v4f*)(W + (size_t)r * 128 + k0 + c4);
      *(v4bf*)(sB + r * LSTR + c4) = cvt4(v);
    }
    __syncthreads();
    v16bf a0 = frag_ld(sA + (wm * 32 + 0)  * LSTR, lane);
    v16bf a1 = frag_ld(sA + (wm * 32 + 16) * LSTR, lane);
    v16bf b0 = frag_ld(sB + (wn * 32 + 0)  * LSTR, lane);
    v16bf b1 = frag_ld(sB + (wn * 32 + 16) * LSTR, lane);
    acc[0][0] = wmma_bf(a0, b0, acc[0][0]);
    acc[0][1] = wmma_bf(a0, b1, acc[0][1]);
    acc[1][0] = wmma_bf(a1, b0, acc[1][0]);
    acc[1][1] = wmma_bf(a1, b1, acc[1][1]);
    __syncthreads();
  }
  const int n = lane & 15, ms = (lane >> 4) << 3;
  for (int i = 0; i < 2; ++i)
    for (int j = 0; j < 2; ++j) {
      int gm = m0 + wm * 32 + i * 16 + ms;
      int gn = wn * 32 + j * 16 + n;
      if (outF) {
        for (int v = 0; v < 8; ++v)
          outF[(size_t)(gm + v) * 128 + gn] = acc[i][j][v];
      }
      if (outT) {
        int batch = gm / rowsPerBatch;
        int ml = gm - batch * rowsPerBatch;
        v8bf t;
        for (int v = 0; v < 8; ++v) t[v] = (__bf16)acc[i][j][v];
        *(v8bf*)(outT + ((size_t)batch * 128 + gn) * rowsPerBatch + ml) = t;
      }
    }
}

// ---------------------------------------------------------------------------
// Kernel 3: fused vertex gather. Per batch:
//   eAtt[e][d] = inc^T @ att ; agg[e][d] = inc^T @ feat   (K = m = 4096)
// A = inc^T (transposed + converted in LDS), B = attT/featT bf16 [d][m].
// ---------------------------------------------------------------------------
__global__ void __launch_bounds__(256) k_vertex_gather(const float* __restrict__ inc,
                                                       const __bf16* __restrict__ attT,
                                                       const __bf16* __restrict__ featT,
                                                       float* __restrict__ eAtt,
                                                       float* __restrict__ agg) {
  __shared__ __align__(16) __bf16 sA[64 * LSTR];    // inc^T tile [64e][32m]
  __shared__ __align__(16) __bf16 sBa[ND * LSTR];   // attT  [128d][32m]
  __shared__ __align__(16) __bf16 sBf[ND * LSTR];   // featT [128d][32m]
  const int tid = threadIdx.x, lane = tid & 31, wave = tid >> 5;
  const int wm = wave >> 2, wn = wave & 3;
  const int b = blockIdx.y;
  const int e0 = blockIdx.x * 64;
  const float*  incB = inc  + (size_t)b * NM * NE;
  const __bf16* aT   = attT + (size_t)b * ND * NM;
  const __bf16* fT   = featT+ (size_t)b * ND * NM;

  v8f cAtt[2][2] = {}, cAgg[2][2] = {};
  for (int k0 = 0; k0 < NM; k0 += 32) {
    if (k0 + 32 < NM) {                           // hint next inc tile -> global_prefetch_b8
      int mm = tid >> 4, e4 = (tid & 15) * 4;
      __builtin_prefetch(incB + (size_t)(k0 + 32 + mm) * NE + e0 + e4, 0, 0);
    }
    for (int i = 0; i < 2; ++i) {                 // transpose-stage inc: [64e][32m]
      int idx = tid + i * 256;
      int mm = idx >> 4, e4 = (idx & 15) * 4;
      v4f v = *(const v4f*)(incB + (size_t)(k0 + mm) * NE + e0 + e4);
      sA[(e4 + 0) * LSTR + mm] = (__bf16)v[0];
      sA[(e4 + 1) * LSTR + mm] = (__bf16)v[1];
      sA[(e4 + 2) * LSTR + mm] = (__bf16)v[2];
      sA[(e4 + 3) * LSTR + mm] = (__bf16)v[3];
    }
    for (int i = 0; i < 2; ++i) {                 // stage both B operands
      int idx = tid + i * 256;
      int r = idx >> 2, c8 = (idx & 3) * 8;
      *(v8bf*)(sBa + r * LSTR + c8) = *(const v8bf*)(aT + (size_t)r * NM + k0 + c8);
      *(v8bf*)(sBf + r * LSTR + c8) = *(const v8bf*)(fT + (size_t)r * NM + k0 + c8);
    }
    __syncthreads();
    v16bf a0  = frag_ld(sA  + (wm * 32 + 0)  * LSTR, lane);
    v16bf a1  = frag_ld(sA  + (wm * 32 + 16) * LSTR, lane);
    v16bf ba0 = frag_ld(sBa + (wn * 32 + 0)  * LSTR, lane);
    v16bf ba1 = frag_ld(sBa + (wn * 32 + 16) * LSTR, lane);
    v16bf bf0 = frag_ld(sBf + (wn * 32 + 0)  * LSTR, lane);
    v16bf bf1 = frag_ld(sBf + (wn * 32 + 16) * LSTR, lane);
    cAtt[0][0] = wmma_bf(a0, ba0, cAtt[0][0]);
    cAtt[0][1] = wmma_bf(a0, ba1, cAtt[0][1]);
    cAtt[1][0] = wmma_bf(a1, ba0, cAtt[1][0]);
    cAtt[1][1] = wmma_bf(a1, ba1, cAtt[1][1]);
    cAgg[0][0] = wmma_bf(a0, bf0, cAgg[0][0]);
    cAgg[0][1] = wmma_bf(a0, bf1, cAgg[0][1]);
    cAgg[1][0] = wmma_bf(a1, bf0, cAgg[1][0]);
    cAgg[1][1] = wmma_bf(a1, bf1, cAgg[1][1]);
    __syncthreads();
  }
  const int n = lane & 15, ms = (lane >> 4) << 3;
  for (int i = 0; i < 2; ++i)
    for (int j = 0; j < 2; ++j) {
      int ge = e0 + wm * 32 + i * 16 + ms;
      int gd = wn * 32 + j * 16 + n;
      for (int v = 0; v < 8; ++v) {
        size_t o = ((size_t)b * NE + ge + v) * ND + gd;
        eAtt[o] = cAtt[i][j][v];
        agg[o]  = cAgg[i][j][v];
      }
    }
}

// ---------------------------------------------------------------------------
// Kernel 4: softmax over e per (b,d) column, fused with agg multiply.
//   sm[b][e][d] = agg[b][e][d] * softmax_e(eAtt[b][:][d])
// ---------------------------------------------------------------------------
__global__ void __launch_bounds__(128) k_softmax_mul(const float* __restrict__ eAtt,
                                                     const float* __restrict__ agg,
                                                     float* __restrict__ sm) {
  const int b = blockIdx.x, d = threadIdx.x;
  const float* x = eAtt + (size_t)b * NE * ND + d;
  const float* g = agg  + (size_t)b * NE * ND + d;
  float*       o = sm   + (size_t)b * NE * ND + d;
  float mx = -3.402823466e38f;
  for (int e = 0; e < NE; ++e) mx = fmaxf(mx, x[(size_t)e * ND]);
  float s = 0.f;
  for (int e = 0; e < NE; ++e) s += __expf(x[(size_t)e * ND] - mx);
  float inv = 1.0f / s;
  for (int e = 0; e < NE; ++e)
    o[(size_t)e * ND] = g[(size_t)e * ND] * __expf(x[(size_t)e * ND] - mx) * inv;
}

// ---------------------------------------------------------------------------
// Kernel 5: edge gather. Per batch: nfpre[m][d] = inc @ ef   (K = e = 2048)
// A = inc (straight convert), B = efT bf16 [d][e].
// ---------------------------------------------------------------------------
__global__ void __launch_bounds__(256) k_edge_gather(const float* __restrict__ inc,
                                                     const __bf16* __restrict__ efT,
                                                     float* __restrict__ nfpre) {
  __shared__ __align__(16) __bf16 sA[64 * LSTR];
  __shared__ __align__(16) __bf16 sB[ND * LSTR];
  const int tid = threadIdx.x, lane = tid & 31, wave = tid >> 5;
  const int wm = wave >> 2, wn = wave & 3;
  const int b = blockIdx.y;
  const int m0 = blockIdx.x * 64;
  const float*  incB = inc + (size_t)b * NM * NE;
  const __bf16* eT   = efT + (size_t)b * ND * NE;

  v8f acc[2][2] = {};
  for (int k0 = 0; k0 < NE; k0 += 32) {
    if (k0 + 32 < NE) {
      int r = tid >> 3, c4 = (tid & 7) * 4;
      __builtin_prefetch(incB + (size_t)(m0 + r) * NE + k0 + 32 + c4, 0, 0);
    }
    for (int i = 0; i < 2; ++i) {                 // A tile [64m][32e], straight
      int idx = tid + i * 256;
      int r = idx >> 3, c4 = (idx & 7) * 4;
      v4f v = *(const v4f*)(incB + (size_t)(m0 + r) * NE + k0 + c4);
      *(v4bf*)(sA + r * LSTR + c4) = cvt4(v);
    }
    for (int i = 0; i < 2; ++i) {                 // B tile [128d][32e]
      int idx = tid + i * 256;
      int r = idx >> 2, c8 = (idx & 3) * 8;
      *(v8bf*)(sB + r * LSTR + c8) = *(const v8bf*)(eT + (size_t)r * NE + k0 + c8);
    }
    __syncthreads();
    v16bf a0 = frag_ld(sA + (wm * 32 + 0)  * LSTR, lane);
    v16bf a1 = frag_ld(sA + (wm * 32 + 16) * LSTR, lane);
    v16bf b0 = frag_ld(sB + (wn * 32 + 0)  * LSTR, lane);
    v16bf b1 = frag_ld(sB + (wn * 32 + 16) * LSTR, lane);
    acc[0][0] = wmma_bf(a0, b0, acc[0][0]);
    acc[0][1] = wmma_bf(a0, b1, acc[0][1]);
    acc[1][0] = wmma_bf(a1, b0, acc[1][0]);
    acc[1][1] = wmma_bf(a1, b1, acc[1][1]);
    __syncthreads();
  }
  const int n = lane & 15, ms = (lane >> 4) << 3;
  for (int i = 0; i < 2; ++i)
    for (int j = 0; j < 2; ++j) {
      int gm = m0 + wm * 32 + i * 16 + ms;
      int gd = wn * 32 + j * 16 + n;
      for (int v = 0; v < 8; ++v)
        nfpre[((size_t)b * NM + gm + v) * ND + gd] = acc[i][j][v];
    }
}

// ---------------------------------------------------------------------------
extern "C" void kernel_launch(void* const* d_in, const int* in_sizes, int n_in,
                              void* d_out, int out_size, void* d_ws, size_t ws_size,
                              hipStream_t stream) {
  const float* features = (const float*)d_in[0];  // [8,4096,128]
  const float* inc      = (const float*)d_in[1];  // [8,4096,2048]
  const float* vc_Wa    = (const float*)d_in[2];  // [128,128]
  const float* vc_Wp    = (const float*)d_in[3];  // [128,128]
  // d_in[4]=vc_alpha, d_in[5]=ec_Wa (dead in reference), d_in[7]=ec_alpha,
  // d_in[8]=n_layers: alpha-blending is a fixed point (x = a*x+(1-a)*x), so the
  // layer loop is the identity — compute each quantity once.
  const float* ec_Wp    = (const float*)d_in[6];  // [128,128]

  float* node_out = (float*)d_out;                        // [8,4096,128]
  float* edge_out = node_out + (size_t)NB * NM * ND;      // [8,2048,128]

  char* w = (char*)d_ws;
  __bf16* attT  = (__bf16*)(w);                            //  8 MB [b][128][4096]
  __bf16* featT = (__bf16*)(w + (8u  << 20));              //  8 MB [b][128][4096]
  float*  eAtt  = (float*) (w + (16u << 20));              //  8 MB [b][2048][128]
  float*  agg   = (float*) (w + (24u << 20));              //  8 MB
  float*  sm    = (float*) (w + (32u << 20));              //  8 MB
  __bf16* efT   = (__bf16*)(w + (40u << 20));              //  4 MB [b][128][2048]
  float*  nfpre = (float*) (w + (44u << 20));              // 16 MB [b][4096][128]

  // 1. featT (bf16, transposed)
  k_transpose_feat<<<dim3(NM / 64, NB), 256, 0, stream>>>(features, featT);
  // 2. att = feat @ Wa^T  -> attT (bf16, transposed)
  k_gemm_nt_128<<<(NB * NM) / 64, 256, 0, stream>>>(features, vc_Wa, nullptr, attT, NM);
  // 3. eAtt = inc^T @ att ; agg = inc^T @ feat    (fused, K=4096)
  k_vertex_gather<<<dim3(NE / 64, NB), 256, 0, stream>>>(inc, attT, featT, eAtt, agg);
  // 4. sm = agg * softmax_e(eAtt)
  k_softmax_mul<<<NB, 128, 0, stream>>>(eAtt, agg, sm);
  // 5. ef = sm @ vc_Wp^T  -> edge_out (f32) and efT (bf16 transposed)
  k_gemm_nt_128<<<(NB * NE) / 64, 256, 0, stream>>>(sm, vc_Wp, edge_out, efT, NE);
  // 6. nfpre = inc @ ef   (K=2048)
  k_edge_gather<<<dim3(NM / 64, NB), 256, 0, stream>>>(inc, efT, nfpre);
  // 7. nf = nfpre @ ec_Wp^T -> node_out
  k_gemm_nt_128<<<(NB * NM) / 64, 256, 0, stream>>>(nfpre, ec_Wp, node_out, nullptr, NM);
}
